// EquivariantUpdate_35150012351086
// MI455X (gfx1250) — compile-verified, matrix operand order checked by reading, not run on hardware
//
#include <hip/hip_runtime.h>
#include <hip/hip_bf16.h>

#define HID 256
#define EPB 128           // edges per block
#define NT  16            // N tiles (16x16) per wave -> 256 cols

typedef __attribute__((ext_vector_type(16))) __bf16 v16bf;
typedef __attribute__((ext_vector_type(8)))  float  v8f;
typedef int v4i_vs __attribute__((vector_size(16)));   // matches builtin param type

#if __has_builtin(__builtin_amdgcn_global_load_async_to_lds_b128) && \
    __has_builtin(__builtin_amdgcn_s_wait_asynccnt)
#define USE_ASYNC_LDS 1
#endif

union U16B { uint4 u; __bf16 b[8]; };

// Load one A/B fragment slice from bf16 memory laid out so that this lane's
// 16 elements are two contiguous 8-element (16B) chunks at p and p+16.
__device__ inline v16bf load_frag_bf16(const __bf16* p) {
    U16B lo, hi;
    lo.u = *reinterpret_cast<const uint4*>(p);
    hi.u = *reinterpret_cast<const uint4*>(p + 16);
    v16bf r;
#pragma unroll
    for (int i = 0; i < 8; ++i) { r[i] = lo.b[i]; r[i + 8] = hi.b[i]; }
    return r;
}

// Build an A fragment slice from f32 memory (gathered activations), converting
// to bf16: cols p[0..7] -> elems 0..7, p[16..23] -> elems 8..15.
__device__ inline v16bf load_frag_f32(const float* p) {
    float4 f0 = *reinterpret_cast<const float4*>(p);
    float4 f1 = *reinterpret_cast<const float4*>(p + 4);
    float4 f2 = *reinterpret_cast<const float4*>(p + 16);
    float4 f3 = *reinterpret_cast<const float4*>(p + 20);
    v16bf r;
    r[0]  = (__bf16)f0.x; r[1]  = (__bf16)f0.y; r[2]  = (__bf16)f0.z; r[3]  = (__bf16)f0.w;
    r[4]  = (__bf16)f1.x; r[5]  = (__bf16)f1.y; r[6]  = (__bf16)f1.z; r[7]  = (__bf16)f1.w;
    r[8]  = (__bf16)f2.x; r[9]  = (__bf16)f2.y; r[10] = (__bf16)f2.z; r[11] = (__bf16)f2.w;
    r[12] = (__bf16)f3.x; r[13] = (__bf16)f3.y; r[14] = (__bf16)f3.z; r[15] = (__bf16)f3.w;
    return r;
}

__device__ inline float silu(float x) { return x / (1.0f + __expf(-x)); }

// Stage one [256 rows x 32 cols] bf16 K-chunk of W^T into LDS.
// 256 threads: thread `tid` copies row `tid` (64 bytes = 4 x b128).
__device__ inline void stage_chunk(const __bf16* __restrict__ gW, int rowStride,
                                   int kOff, __bf16* ldsBuf, int tid) {
    const __bf16* src = gW + (size_t)tid * rowStride + kOff;
    __bf16* dst = ldsBuf + tid * 32;
#ifdef USE_ASYNC_LDS
    __builtin_amdgcn_global_load_async_to_lds_b128((v4i_vs*)(src),      (v4i_vs*)(dst),      0, 0);
    __builtin_amdgcn_global_load_async_to_lds_b128((v4i_vs*)(src + 8),  (v4i_vs*)(dst + 8),  0, 0);
    __builtin_amdgcn_global_load_async_to_lds_b128((v4i_vs*)(src + 16), (v4i_vs*)(dst + 16), 0, 0);
    __builtin_amdgcn_global_load_async_to_lds_b128((v4i_vs*)(src + 24), (v4i_vs*)(dst + 24), 0, 0);
#else
    const uint4* s4 = reinterpret_cast<const uint4*>(src);
    uint4* d4 = reinterpret_cast<uint4*>(dst);
    d4[0] = s4[0]; d4[1] = s4[1]; d4[2] = s4[2]; d4[3] = s4[3];
#endif
}

__device__ inline void stage_fence() {
#ifdef USE_ASYNC_LDS
    __builtin_amdgcn_s_wait_asynccnt(0);
#endif
    __syncthreads();
}

// ---------------------------------------------------------------------------
// prep: transpose+convert weights to bf16 W^T, zero dist_agg, copy pos->out.
// ---------------------------------------------------------------------------
__global__ void prep_kernel(const float* __restrict__ pos,
                            const float* __restrict__ cm_W1,   // [768,256]
                            const float* __restrict__ cm_W2,   // [256,256]
                            float* __restrict__ out_pos,       // [10000,3]
                            float* __restrict__ dist_agg,      // [10000]
                            __bf16* __restrict__ W1t,          // [256,768]
                            __bf16* __restrict__ W2t,          // [256,256]
                            int nNodes) {
    int id = blockIdx.x * blockDim.x + threadIdx.x;
    if (id < 768 * 256) {
        int n = id / 768, k = id - n * 768;
        W1t[n * 768 + k] = (__bf16)cm_W1[k * 256 + n];
    }
    if (id < 256 * 256) {
        int n = id >> 8, k = id & 255;
        W2t[n * 256 + k] = (__bf16)cm_W2[k * 256 + n];
    }
    if (id < nNodes * 3) out_pos[id] = pos[id];
    if (id < nNodes)     dist_agg[id] = 0.0f;
}

// ---------------------------------------------------------------------------
// edge kernel: 3-layer MLP on each edge via bf16 WMMA + scatter-add.
// Block = 256 threads = 8 waves; wave w owns edges [blk*128 + w*16, +16).
// Weights are staged in LDS in double-buffered 32-wide K-chunks (async copy).
// Dynamic LDS: X1[128*256] bf16 (64KB) | Wbuf0 (16KB) | Wbuf1 (16KB) = 96KB.
// ---------------------------------------------------------------------------
__global__ void __launch_bounds__(256)
edge_kernel(const float* __restrict__ h,            // [N,256]
            const int*   __restrict__ edge_index,   // [2,E]
            const float* __restrict__ coord_diff,   // [E,3]
            const float* __restrict__ distances,    // [E,1]
            const float* __restrict__ edge_attr,    // [E,256]
            const __bf16* __restrict__ W1t,         // [256,768] (W1^T)
            const float* __restrict__ b1,           // [256]
            const __bf16* __restrict__ W2t,         // [256,256] (W2^T)
            const float* __restrict__ b2,           // [256]
            const float* __restrict__ W3,           // [256] (cm_W3[:,0])
            const float* __restrict__ b3v,          // [1]
            float* __restrict__ out_pos,            // [N,3]
            float* __restrict__ dist_agg,           // [N]
            int nEdges) {
    extern __shared__ char smem[];
    __bf16* X1    = (__bf16*)smem;                   // 64 KB
    __bf16* Wbuf0 = (__bf16*)(smem + 65536);         // 16 KB
    __bf16* Wbuf1 = (__bf16*)(smem + 65536 + 16384); // 16 KB

    const int tid  = threadIdx.x;
    const int wave = tid >> 5;
    const int lane = tid & 31;
    const int m    = lane & 15;       // M row within strip / N col within tile
    const int half = lane >> 4;       // K-half selector
    const int khb  = half * 8;
    const int eBase = blockIdx.x * EPB;

    // distances scatter (one thread per edge of this block)
    if (tid < EPB) {
        const int e = eBase + tid;
        atomicAdd(&dist_agg[edge_index[e]], distances[e]);
    }

    const int eRow = eBase + wave * 16 + m;       // this lane's A-matrix row
    const int iie  = edge_index[eRow];
    const int jje  = edge_index[nEdges + eRow];

    const float* srcH0 = h + (size_t)iie * HID;
    const float* srcH1 = h + (size_t)jje * HID;
    const float* srcEA = edge_attr + (size_t)eRow * HID;
    __builtin_prefetch(srcEA, 0, 1);   // global_prefetch_b8 on streaming source

    const v8f vzero = {0.f, 0.f, 0.f, 0.f, 0.f, 0.f, 0.f, 0.f};
    v8f acc[NT];
#pragma unroll
    for (int t = 0; t < NT; ++t) acc[t] = vzero;

    // ------------------- layer 1: K = 768 (24 chunks of 32) -----------------
    stage_chunk(W1t, 768, 0, Wbuf0, tid);
    stage_fence();
    for (int c = 0; c < 24; ++c) {
        __bf16* cur = (c & 1) ? Wbuf1 : Wbuf0;
        __bf16* nxt = (c & 1) ? Wbuf0 : Wbuf1;
        if (c + 1 < 24) stage_chunk(W1t, 768, (c + 1) * 32, nxt, tid);

        const int s  = c >> 3;                 // which gathered source
        const int k0 = (c & 7) * 32;           // col offset within source
        const float* base = (s == 0) ? srcH0 : (s == 1) ? srcH1 : srcEA;
        const v16bf a = load_frag_f32(base + k0 + khb);
#pragma unroll
        for (int t = 0; t < NT; ++t) {
            const __bf16* wp = cur + (t * 16 + m) * 32 + khb;
            const v16bf bfrag = load_frag_bf16(wp);
            acc[t] = __builtin_amdgcn_wmma_f32_16x16x32_bf16(
                false, a, false, bfrag, (short)0, acc[t], false, false);
        }
        stage_fence();   // chunk c+1 staged everywhere before next iteration
    }

    // bias + silu -> LDS (C layout: lane = col within tile, VGPR r = row)
#pragma unroll
    for (int t = 0; t < NT; ++t) {
        const int col   = t * 16 + m;
        const float bia = b1[col];
#pragma unroll
        for (int r = 0; r < 8; ++r) {
            const float x = silu(acc[t][r] + bia);
            X1[(wave * 16 + r + half * 8) * HID + col] = (__bf16)x;
        }
        acc[t] = vzero;
    }

    // ------------------- layer 2: K = 256 (8 chunks of 32) ------------------
    stage_chunk(W2t, HID, 0, Wbuf0, tid);
    stage_fence();
    const __bf16* xrow = X1 + (size_t)(wave * 16 + m) * HID;
    for (int c = 0; c < 8; ++c) {
        __bf16* cur = (c & 1) ? Wbuf1 : Wbuf0;
        __bf16* nxt = (c & 1) ? Wbuf0 : Wbuf1;
        if (c + 1 < 8) stage_chunk(W2t, HID, (c + 1) * 32, nxt, tid);

        const int k0 = c * 32;
        const v16bf a = load_frag_bf16(xrow + k0 + khb);
#pragma unroll
        for (int t = 0; t < NT; ++t) {
            const __bf16* wp = cur + (t * 16 + m) * 32 + khb;
            const v16bf bfrag = load_frag_bf16(wp);
            acc[t] = __builtin_amdgcn_wmma_f32_16x16x32_bf16(
                false, a, false, bfrag, (short)0, acc[t], false, false);
        }
        stage_fence();
    }

    // ------------------- layer 3: bias+silu then 256->1 dot ------------------
    float part[8];
#pragma unroll
    for (int r = 0; r < 8; ++r) part[r] = 0.0f;
#pragma unroll
    for (int t = 0; t < NT; ++t) {
        const int col   = t * 16 + m;
        const float bia = b2[col];
        const float w3  = W3[col];
#pragma unroll
        for (int r = 0; r < 8; ++r)
            part[r] += silu(acc[t][r] + bia) * w3;
    }
#pragma unroll
    for (int r = 0; r < 8; ++r) {
#pragma unroll
        for (int off = 8; off >= 1; off >>= 1)
            part[r] += __shfl_xor(part[r], off, 32);
    }

    const float b3 = b3v[0];
    // rows wave*16 + r + half*8 ; lanes m<3 scatter the 3 coords
#pragma unroll
    for (int r = 0; r < 8; ++r) {
        if (m < 3) {
            const int e   = eBase + wave * 16 + r + half * 8;
            const float phi = part[r] + b3;
            const int dst = edge_index[e];
            atomicAdd(&out_pos[dst * 3 + m], coord_diff[(size_t)e * 3 + m] * phi);
        }
    }
}

// ---------------------------------------------------------------------------
// node kernel: h_new = h + MLP_{1->16->256}(dist_agg)
// ---------------------------------------------------------------------------
__global__ void node_kernel(const float* __restrict__ h,
                            const float* __restrict__ dist_agg,
                            const float* __restrict__ deW1,  // [1,16]
                            const float* __restrict__ deb1,  // [16]
                            const float* __restrict__ deW2,  // [16,256]
                            const float* __restrict__ deb2,  // [256]
                            float* __restrict__ out_h) {
    const int n = blockIdx.x;
    const int c = threadIdx.x;   // 256 threads, one column each
    const float d = dist_agg[n];
    float acc = deb2[c];
#pragma unroll
    for (int j = 0; j < 16; ++j) {
        const float t = silu(d * deW1[j] + deb1[j]);
        acc += t * deW2[j * 256 + c];
    }
    out_h[(size_t)n * 256 + c] = h[(size_t)n * 256 + c] + acc;
}

// ---------------------------------------------------------------------------
extern "C" void kernel_launch(void* const* d_in, const int* in_sizes, int n_in,
                              void* d_out, int out_size, void* d_ws, size_t ws_size,
                              hipStream_t stream) {
    const float* h          = (const float*)d_in[0];
    const float* pos        = (const float*)d_in[1];
    const int*   edge_index = (const int*)  d_in[2];
    const float* coord_diff = (const float*)d_in[3];
    const float* distances  = (const float*)d_in[4];
    const float* edge_attr  = (const float*)d_in[5];
    const float* de_W1      = (const float*)d_in[6];
    const float* de_b1      = (const float*)d_in[7];
    const float* de_W2      = (const float*)d_in[8];
    const float* de_b2      = (const float*)d_in[9];
    const float* cm_W1      = (const float*)d_in[10];
    const float* cm_b1      = (const float*)d_in[11];
    const float* cm_W2      = (const float*)d_in[12];
    const float* cm_b2      = (const float*)d_in[13];
    const float* cm_W3      = (const float*)d_in[14];
    const float* cm_b3      = (const float*)d_in[15];

    const int E = in_sizes[4];          // 320000
    const int N = in_sizes[1] / 3;      // 10000

    float* out_pos = (float*)d_out;           // [N,3]
    float* out_h   = (float*)d_out + N * 3;   // [N,256]

    // workspace layout
    char* ws = (char*)d_ws;
    __bf16* W1t      = (__bf16*)(ws);                      // 256*768*2 = 393216 B
    __bf16* W2t      = (__bf16*)(ws + 393216);             // 256*256*2 = 131072 B
    float*  dist_agg = (float*) (ws + 393216 + 131072);    // N*4 B

    prep_kernel<<<768, 256, 0, stream>>>(pos, cm_W1, cm_W2,
                                         out_pos, dist_agg, W1t, W2t, N);

    const size_t smem_bytes = 65536 + 2 * 16384;   // 96 KB dynamic LDS
    edge_kernel<<<E / EPB, 256, smem_bytes, stream>>>(
        h, edge_index, coord_diff, distances, edge_attr,
        W1t, cm_b1, W2t, cm_b2, cm_W3, cm_b3, out_pos, dist_agg, E);

    node_kernel<<<N, 256, 0, stream>>>(h, dist_agg, de_W1, de_b1, de_W2, de_b2,
                                       out_h);
}
